// PointNetPlusPlus_33114197852812
// MI455X (gfx1250) — matrix operand order repaired and checked by reading, not compiled, not run on hardware
//
#include <hip/hip_runtime.h>
#include <hip/hip_bf16.h>

typedef __attribute__((ext_vector_type(16))) _Float16 v16h;
typedef __attribute__((ext_vector_type(8)))  _Float16 v8h;
typedef __attribute__((ext_vector_type(8)))  float    v8f;

#define BATCH 8
#define NPTS  4096
#define N1    2048
#define N2    512
#define KNB   64

// ================= WMMA helpers (CDNA5 16x16x32 f16 -> f32) =================
// Compute h^T = W^T x act^T :
//   A operand = W^T tile (16 rows = out-channels n, 32 cols = k), pre-swizzled
//   B operand = activation tile (32 rows = k, 16 cols = neighbors/points m),
//               read row-major from LDS act[m][k] (contiguous per lane)
//   D tile: lane L <-> m = mbase + L%16 ; vgpr r <-> n = nbase + r + (L/16)*8

// A (weights) pre-swizzled in global: one aligned 32B load per lane
__device__ __forceinline__ v16h load_w_packed(const _Float16* wp, int ksteps,
                                              int nt, int ks, int lane) {
  return *(const v16h*)(wp + ((((size_t)nt * ksteps + ks) * 32 + lane) << 4));
}

// B (activations) from LDS row-major [rows=m][ld]: 16 contiguous halves/lane
__device__ __forceinline__ v16h load_act(const _Float16* s, int ld, int mbase,
                                         int kbase, int lane) {
  const _Float16* p = s + (mbase + (lane & 15)) * ld + kbase + ((lane >> 4) << 4);
  v16h b;
#pragma unroll
  for (int h = 0; h < 16; ++h) b[h] = p[h];
  return b;
}

// bias: r <-> consecutive channels -> contiguous 32B load
__device__ __forceinline__ v8f bias_acc(const float* bias, int nbase, int lane) {
  const float* bp = bias + nbase + ((lane >> 4) << 3);
  v8f acc;
#pragma unroll
  for (int r = 0; r < 8; ++r) acc[r] = bp[r];
  return acc;
}

__device__ __forceinline__ v8f wmma_f16(v16h a, v16h b, v8f c) {
  return __builtin_amdgcn_wmma_f32_16x16x32_f16(false, a, false, b, (short)0, c,
                                                false, false);
}

// ReLU + store D tile to LDS h[m][n]: 8 consecutive channels -> 1 ds_store_b128
__device__ __forceinline__ void store_relu_row(_Float16* s, int ld, int mbase,
                                               int nbase, v8f acc, int lane) {
  int m = mbase + (lane & 15);
  int n0 = nbase + ((lane >> 4) << 3);
  v8h pk;
#pragma unroll
  for (int r = 0; r < 8; ++r) pk[r] = (_Float16)fmaxf(acc[r], 0.f);
  *(v8h*)&s[m * ld + n0] = pk;
}

// lane-reduction max over 16 neighbors per half, then 16B store (f16)
__device__ __forceinline__ void maxpool_store_f16(_Float16* xo, int nbase,
                                                  v8f rmax, int lane) {
#pragma unroll
  for (int r = 0; r < 8; ++r) {
    float v = rmax[r];
    v = fmaxf(v, __shfl_xor(v, 1, 32));
    v = fmaxf(v, __shfl_xor(v, 2, 32));
    v = fmaxf(v, __shfl_xor(v, 4, 32));
    v = fmaxf(v, __shfl_xor(v, 8, 32));
    rmax[r] = v;
  }
  if ((lane & 15) == 0) {
    v8h pk;
#pragma unroll
    for (int r = 0; r < 8; ++r) pk[r] = (_Float16)rmax[r];
    *(v8h*)&xo[nbase + ((lane >> 4) << 3)] = pk;
  }
}

__device__ __forceinline__ void maxpool_store_f32(float* po, int nbase,
                                                  v8f rmax, int lane) {
#pragma unroll
  for (int r = 0; r < 8; ++r) {
    float v = rmax[r];
    v = fmaxf(v, __shfl_xor(v, 1, 32));
    v = fmaxf(v, __shfl_xor(v, 2, 32));
    v = fmaxf(v, __shfl_xor(v, 4, 32));
    v = fmaxf(v, __shfl_xor(v, 8, 32));
    rmax[r] = v;
  }
  if ((lane & 15) == 0) {
    int n0 = nbase + ((lane >> 4) << 3);
    *(float4*)&po[n0] = make_float4(rmax[0], rmax[1], rmax[2], rmax[3]);
    *(float4*)&po[n0 + 4] = make_float4(rmax[4], rmax[5], rmax[6], rmax[7]);
  }
}

// ======= weight pack: f32 [Kin][Nout] -> A-operand tiles of W^T ==============
// out[((nt*ksteps+ks)*32 + lane)*16 + h] :
//   n = nt*16 + lane%16 ; k = ks*32 + (lane/16)*16 + h
__global__ void pack_w_kernel(const float* __restrict__ w, int Kin, int Nout,
                              _Float16* __restrict__ out) {
  int ksteps = (Kin + 31) >> 5;
  int total = (Nout >> 4) * ksteps * 32 * 16;
  int i = blockIdx.x * blockDim.x + threadIdx.x;
  if (i >= total) return;
  int h = i & 15;
  int lane = (i >> 4) & 31;
  int t = i >> 9;  // nt*ksteps + ks
  int ks = t % ksteps, nt = t / ksteps;
  int n = nt * 16 + (lane & 15);
  int k = ks * 32 + ((lane >> 4) << 4) + h;
  out[i] = (k < Kin) ? (_Float16)w[(size_t)k * Nout + n] : (_Float16)0.f;
}

// ---------------- Farthest point sampling: one block per cloud ----------------
__global__ void fps_kernel(const float* __restrict__ pts, int n, int nsamp,
                           int* __restrict__ idx_out, float* __restrict__ cent_out) {
  int b = blockIdx.x;
  const float* p = pts + (size_t)b * n * 3;
  int* idxo = idx_out + (size_t)b * nsamp;
  float* co = cent_out + (size_t)b * nsamp * 3;
  __shared__ float mind[NPTS];
  __shared__ float rv[256];
  __shared__ int   ri[256];
  __shared__ float curp[3];
  int tid = threadIdx.x;
  if (tid == 0) {
    idxo[0] = 0;
    curp[0] = p[0]; curp[1] = p[1]; curp[2] = p[2];
    co[0] = p[0]; co[1] = p[1]; co[2] = p[2];
  }
  __syncthreads();
  float cx = curp[0], cy = curp[1], cz = curp[2];
  for (int i = tid; i < n; i += blockDim.x) {
    float dx = p[3 * i] - cx, dy = p[3 * i + 1] - cy, dz = p[3 * i + 2] - cz;
    mind[i] = dx * dx + dy * dy + dz * dz;
  }
  __syncthreads();
  for (int s = 1; s < nsamp; ++s) {
    float bv = -1.f; int bi = 0;
    for (int i = tid; i < n; i += blockDim.x) {
      float v = mind[i];
      if (v > bv) { bv = v; bi = i; }
    }
    rv[tid] = bv; ri[tid] = bi;
    __syncthreads();
    for (int st = 128; st > 0; st >>= 1) {
      if (tid < st) {
        float ov = rv[tid + st]; int oi = ri[tid + st];
        if (ov > rv[tid] || (ov == rv[tid] && oi < ri[tid])) { rv[tid] = ov; ri[tid] = oi; }
      }
      __syncthreads();
    }
    if (tid == 0) {
      int nx = ri[0];
      idxo[s] = nx;
      curp[0] = p[3 * nx]; curp[1] = p[3 * nx + 1]; curp[2] = p[3 * nx + 2];
      co[3 * s] = curp[0]; co[3 * s + 1] = curp[1]; co[3 * s + 2] = curp[2];
    }
    __syncthreads();
    cx = curp[0]; cy = curp[1]; cz = curp[2];
    for (int i = tid; i < n; i += blockDim.x) {
      float dx = p[3 * i] - cx, dy = p[3 * i + 1] - cy, dz = p[3 * i + 2] - cz;
      float d = dx * dx + dy * dy + dz * dz;
      if (d < mind[i]) mind[i] = d;
    }
    __syncthreads();
  }
}

// ---------------- Ball query: K nearest within radius, one thread per center ----
__global__ void ballq_kernel(const float* __restrict__ cent,
                             const float* __restrict__ pts, int M, int n, float r2,
                             int* __restrict__ nbr,
                             unsigned long long* __restrict__ maskw) {
  __shared__ float sd[64][KNB];
  __shared__ int   si[64][KNB];
  int g = blockIdx.x * 64 + threadIdx.x;
  int b = g / M, ci = g % M;
  const float* p = pts + (size_t)b * n * 3;
  const float* c = cent + (size_t)(b * M + ci) * 3;
  float cx = c[0], cy = c[1], cz = c[2];
  float* dl = sd[threadIdx.x];
  int*   il = si[threadIdx.x];
  int cnt = 0;
  for (int j = 0; j < n; ++j) {
    float dx = p[3 * j] - cx, dy = p[3 * j + 1] - cy, dz = p[3 * j + 2] - cz;
    float d2 = dx * dx + dy * dy + dz * dz;
    if (d2 > r2) continue;
    if (cnt == KNB && d2 >= dl[KNB - 1]) continue;
    int pos = (cnt < KNB) ? cnt : (KNB - 1);
    while (pos > 0 && dl[pos - 1] > d2) {  // stable: equal keeps earlier index
      dl[pos] = dl[pos - 1]; il[pos] = il[pos - 1]; --pos;
    }
    dl[pos] = d2; il[pos] = j;
    if (cnt < KNB) ++cnt;
  }
  int* no = nbr + (size_t)g * KNB;
  for (int k = 0; k < KNB; ++k) no[k] = (k < cnt) ? il[k] : 0;
  maskw[g] = (cnt >= 64) ? ~0ull : ((1ull << cnt) - 1ull);
}

// ---------------- SA1: rel-pos MLP 3->64->64 + masked max. 1 wave / center ----
__global__ void sa1_kernel(const float* __restrict__ pos, const float* __restrict__ c1,
                           const int* __restrict__ nbr,
                           const unsigned long long* __restrict__ maskw,
                           const _Float16* __restrict__ wpa, const float* __restrict__ b1a,
                           const _Float16* __restrict__ wpb, const float* __restrict__ b1b,
                           _Float16* __restrict__ x1) {
  int g = blockIdx.x;  // b*N1 + ci
  int b = g >> 11;
  int lane = threadIdx.x;
  __shared__ _Float16 feat[KNB][32];   // [64 nbrs][3 pad 32]
  __shared__ _Float16 h1[KNB][64];
  __builtin_prefetch(wpa, 0, 1);
  __builtin_prefetch(wpb, 0, 1);
  const float* p = pos + (size_t)b * NPTS * 3;
  const float* c = c1 + (size_t)g * 3;
  const int* nb = nbr + (size_t)g * KNB;
  unsigned long long mk = maskw[g];
  float cx = c[0], cy = c[1], cz = c[2];
  for (int t = lane; t < KNB; t += 32) {
    uint4* fr = (uint4*)&feat[t][0];
    fr[0] = make_uint4(0, 0, 0, 0); fr[1] = make_uint4(0, 0, 0, 0);
    fr[2] = make_uint4(0, 0, 0, 0); fr[3] = make_uint4(0, 0, 0, 0);
    int j = nb[t];
    feat[t][0] = (_Float16)(p[3 * j] - cx);
    feat[t][1] = (_Float16)(p[3 * j + 1] - cy);
    feat[t][2] = (_Float16)(p[3 * j + 2] - cz);
  }
  __syncthreads();
  // layer A: 3->64 (ksteps=1)
  for (int nt = 0; nt < 4; ++nt) {
    v16h wa = load_w_packed(wpa, 1, nt, 0, lane);
    for (int mt = 0; mt < 4; ++mt) {
      v8f acc = bias_acc(b1a, nt * 16, lane);
      acc = wmma_f16(wa, load_act(&feat[0][0], 32, mt * 16, 0, lane), acc);
      store_relu_row(&h1[0][0], 64, mt * 16, nt * 16, acc, lane);
    }
  }
  __syncthreads();
  // layer B: 64->64 + relu + masked max over neighbors (ksteps=2)
  _Float16* xo = x1 + (size_t)g * 64;
  for (int nt = 0; nt < 4; ++nt) {
    v16h wa0 = load_w_packed(wpb, 2, nt, 0, lane);
    v16h wa1 = load_w_packed(wpb, 2, nt, 1, lane);
    v8f rmax;
#pragma unroll
    for (int r = 0; r < 8; ++r) rmax[r] = -INFINITY;
    for (int mt = 0; mt < 4; ++mt) {
      v8f acc = bias_acc(b1b, nt * 16, lane);
      acc = wmma_f16(wa0, load_act(&h1[0][0], 64, mt * 16, 0, lane), acc);
      acc = wmma_f16(wa1, load_act(&h1[0][0], 64, mt * 16, 32, lane), acc);
      bool valid = (mk >> (mt * 16 + (lane & 15))) & 1ull;
#pragma unroll
      for (int r = 0; r < 8; ++r)
        rmax[r] = fmaxf(rmax[r], valid ? fmaxf(acc[r], 0.f) : -INFINITY);
    }
    maxpool_store_f16(xo, nt * 16, rmax, lane);
  }
}

// ---------------- SA2: [x1_j, rel] MLP 67->128->128 + masked max -------------
__global__ void sa2_kernel(const float* __restrict__ c1, const float* __restrict__ c2,
                           const int* __restrict__ nbr,
                           const unsigned long long* __restrict__ maskw,
                           const _Float16* __restrict__ x1,
                           const _Float16* __restrict__ wpa, const float* __restrict__ b2a,
                           const _Float16* __restrict__ wpb, const float* __restrict__ b2b,
                           _Float16* __restrict__ x2) {
  int g = blockIdx.x;  // b*N2 + ci
  int b = g / N2;
  int lane = threadIdx.x;
  __shared__ _Float16 feat[KNB][96];   // [64 nbrs][67 pad 96]
  __shared__ _Float16 h1[KNB][128];
  __builtin_prefetch(wpa, 0, 1);
  __builtin_prefetch(wpb, 0, 1);
  const float* pp = c1 + (size_t)b * N1 * 3;
  const float* c = c2 + (size_t)g * 3;
  const int* nb = nbr + (size_t)g * KNB;
  unsigned long long mk = maskw[g];
  float cx = c[0], cy = c[1], cz = c[2];
  for (int t = lane; t < KNB; t += 32) {
    int j = nb[t];
    const uint4* xf = (const uint4*)(x1 + ((size_t)b * N1 + j) * 64);
    uint4* fr = (uint4*)&feat[t][0];  // rows 192B apart -> 16B aligned
#pragma unroll
    for (int k = 0; k < 4; ++k) fr[k] = xf[k];
    uint4* ft = (uint4*)&feat[t][64];
#pragma unroll
    for (int k = 0; k < 4; ++k) ft[k] = make_uint4(0, 0, 0, 0);
    feat[t][64] = (_Float16)(pp[3 * j] - cx);
    feat[t][65] = (_Float16)(pp[3 * j + 1] - cy);
    feat[t][66] = (_Float16)(pp[3 * j + 2] - cz);
  }
  __syncthreads();
  // layer A: 67->128 (ksteps=3, zero-padded)
  for (int nt = 0; nt < 8; ++nt) {
    v16h wa[3];
#pragma unroll
    for (int ks = 0; ks < 3; ++ks) wa[ks] = load_w_packed(wpa, 3, nt, ks, lane);
    for (int mt = 0; mt < 4; ++mt) {
      v8f acc = bias_acc(b2a, nt * 16, lane);
#pragma unroll
      for (int ks = 0; ks < 3; ++ks)
        acc = wmma_f16(wa[ks], load_act(&feat[0][0], 96, mt * 16, ks * 32, lane), acc);
      store_relu_row(&h1[0][0], 128, mt * 16, nt * 16, acc, lane);
    }
  }
  __syncthreads();
  // layer B: 128->128 + relu + masked max (ksteps=4)
  _Float16* xo = x2 + (size_t)g * 128;
  for (int nt = 0; nt < 8; ++nt) {
    v16h wa[4];
#pragma unroll
    for (int ks = 0; ks < 4; ++ks) wa[ks] = load_w_packed(wpb, 4, nt, ks, lane);
    v8f rmax;
#pragma unroll
    for (int r = 0; r < 8; ++r) rmax[r] = -INFINITY;
    for (int mt = 0; mt < 4; ++mt) {
      v8f acc = bias_acc(b2b, nt * 16, lane);
#pragma unroll
      for (int ks = 0; ks < 4; ++ks)
        acc = wmma_f16(wa[ks], load_act(&h1[0][0], 128, mt * 16, ks * 32, lane), acc);
      bool valid = (mk >> (mt * 16 + (lane & 15))) & 1ull;
#pragma unroll
      for (int r = 0; r < 8; ++r)
        rmax[r] = fmaxf(rmax[r], valid ? fmaxf(acc[r], 0.f) : -INFINITY);
    }
    maxpool_store_f16(xo, nt * 16, rmax, lane);
  }
}

// ---------------- SA3: [x2, c2] MLP 131->256->1024, chunk-max ----------------
__global__ void sa3_kernel(const float* __restrict__ c2, const _Float16* __restrict__ x2,
                           const _Float16* __restrict__ wpa, const float* __restrict__ b3a,
                           const _Float16* __restrict__ wpb, const float* __restrict__ b3b,
                           float* __restrict__ partial) {
  int b = blockIdx.x >> 3;
  int chunk = blockIdx.x & 7;  // 64 rows per chunk
  int lane = threadIdx.x;
  __shared__ _Float16 fin[64][160];  // [rows][131 pad 160]
  __shared__ _Float16 h1[64][256];
  __builtin_prefetch(wpa, 0, 1);
  __builtin_prefetch(wpb, 0, 1);
  for (int t = lane; t < 64; t += 32) {
    int pt = chunk * 64 + t;
    const uint4* xf = (const uint4*)(x2 + ((size_t)b * N2 + pt) * 128);
    const float* c = c2 + ((size_t)b * N2 + pt) * 3;
    uint4* fr = (uint4*)&fin[t][0];  // rows 320B apart -> 16B aligned
#pragma unroll
    for (int k = 0; k < 8; ++k) fr[k] = xf[k];
    uint4* ft = (uint4*)&fin[t][128];
#pragma unroll
    for (int k = 0; k < 2; ++k) ft[k] = make_uint4(0, 0, 0, 0);
    fin[t][128] = (_Float16)c[0];
    fin[t][129] = (_Float16)c[1];
    fin[t][130] = (_Float16)c[2];
  }
  __syncthreads();
  // layer A: 131->256 + relu (ksteps=5, zero-padded)
  for (int nt = 0; nt < 16; ++nt) {
    v16h wa[5];
#pragma unroll
    for (int ks = 0; ks < 5; ++ks) wa[ks] = load_w_packed(wpa, 5, nt, ks, lane);
    for (int mt = 0; mt < 4; ++mt) {
      v8f acc = bias_acc(b3a, nt * 16, lane);
#pragma unroll
      for (int ks = 0; ks < 5; ++ks)
        acc = wmma_f16(wa[ks], load_act(&fin[0][0], 160, mt * 16, ks * 32, lane), acc);
      store_relu_row(&h1[0][0], 256, mt * 16, nt * 16, acc, lane);
    }
  }
  __syncthreads();
  // layer B: 256->1024, no relu, max over the 64 rows (ksteps=8)
  float* po = partial + ((size_t)b * 8 + chunk) * 1024;
  for (int nt = 0; nt < 64; ++nt) {
    v16h wa[8];
#pragma unroll
    for (int ks = 0; ks < 8; ++ks) wa[ks] = load_w_packed(wpb, 8, nt, ks, lane);
    v8f rmax;
#pragma unroll
    for (int r = 0; r < 8; ++r) rmax[r] = -INFINITY;
    for (int mt = 0; mt < 4; ++mt) {
      v8f acc = bias_acc(b3b, nt * 16, lane);
#pragma unroll
      for (int ks = 0; ks < 8; ++ks)
        acc = wmma_f16(wa[ks], load_act(&h1[0][0], 256, mt * 16, ks * 32, lane), acc);
#pragma unroll
      for (int r = 0; r < 8; ++r) rmax[r] = fmaxf(rmax[r], acc[r]);
    }
    maxpool_store_f32(po, nt * 16, rmax, lane);
  }
}

// ---------------- final max over chunks -> d_out [B,1024] --------------------
__global__ void reduce_max_kernel(const float* __restrict__ part,
                                  float* __restrict__ out) {
  int i = blockIdx.x * blockDim.x + threadIdx.x;
  if (i >= BATCH * 1024) return;
  int b = i >> 10, j = i & 1023;
  float m = -INFINITY;
  for (int c = 0; c < 8; ++c) m = fmaxf(m, part[((size_t)b * 8 + c) * 1024 + j]);
  out[i] = m;
}

// ---------------- host launcher ----------------
extern "C" void kernel_launch(void* const* d_in, const int* in_sizes, int n_in,
                              void* d_out, int out_size, void* d_ws, size_t ws_size,
                              hipStream_t stream) {
  (void)in_sizes; (void)n_in; (void)out_size; (void)ws_size;
  const float* pos = (const float*)d_in[0];
  const float* w1a = (const float*)d_in[1];  const float* b1a = (const float*)d_in[2];
  const float* w1b = (const float*)d_in[3];  const float* b1b = (const float*)d_in[4];
  const float* w2a = (const float*)d_in[5];  const float* b2a = (const float*)d_in[6];
  const float* w2b = (const float*)d_in[7];  const float* b2b = (const float*)d_in[8];
  const float* w3a = (const float*)d_in[9];  const float* b3a = (const float*)d_in[10];
  const float* w3b = (const float*)d_in[11]; const float* b3b = (const float*)d_in[12];
  float* out = (float*)d_out;

  char* ws = (char*)d_ws;
  size_t off = 0;
  auto alloc = [&](size_t bytes) -> char* {
    char* p = ws + off;
    off += (bytes + 255) & ~(size_t)255;
    return p;
  };
  _Float16* w1a16 = (_Float16*)alloc((size_t)4 * 1 * 512 * 2);
  _Float16* w1b16 = (_Float16*)alloc((size_t)4 * 2 * 512 * 2);
  _Float16* w2a16 = (_Float16*)alloc((size_t)8 * 3 * 512 * 2);
  _Float16* w2b16 = (_Float16*)alloc((size_t)8 * 4 * 512 * 2);
  _Float16* w3a16 = (_Float16*)alloc((size_t)16 * 5 * 512 * 2);
  _Float16* w3b16 = (_Float16*)alloc((size_t)64 * 8 * 512 * 2);
  int* idx1 = (int*)alloc((size_t)BATCH * N1 * 4);
  float* c1 = (float*)alloc((size_t)BATCH * N1 * 3 * 4);
  int* nbr1 = (int*)alloc((size_t)BATCH * N1 * KNB * 4);
  unsigned long long* mask1 = (unsigned long long*)alloc((size_t)BATCH * N1 * 8);
  _Float16* x1 = (_Float16*)alloc((size_t)BATCH * N1 * 64 * 2);
  int* idx2 = (int*)alloc((size_t)BATCH * N2 * 4);
  float* c2 = (float*)alloc((size_t)BATCH * N2 * 3 * 4);
  int* nbr2 = (int*)alloc((size_t)BATCH * N2 * KNB * 4);
  unsigned long long* mask2 = (unsigned long long*)alloc((size_t)BATCH * N2 * 8);
  _Float16* x2 = (_Float16*)alloc((size_t)BATCH * N2 * 128 * 2);
  float* partial = (float*)alloc((size_t)BATCH * 8 * 1024 * 4);

  auto pack = [&](const float* s, int Kin, int Nout, _Float16* d) {
    int total = (Nout >> 4) * ((Kin + 31) >> 5) * 512;
    pack_w_kernel<<<(total + 255) / 256, 256, 0, stream>>>(s, Kin, Nout, d);
  };
  pack(w1a, 3, 64, w1a16);
  pack(w1b, 64, 64, w1b16);
  pack(w2a, 67, 128, w2a16);
  pack(w2b, 128, 128, w2b16);
  pack(w3a, 131, 256, w3a16);
  pack(w3b, 256, 1024, w3b16);

  // SA1
  fps_kernel<<<BATCH, 256, 0, stream>>>(pos, NPTS, N1, idx1, c1);
  ballq_kernel<<<BATCH * N1 / 64, 64, 0, stream>>>(c1, pos, N1, NPTS, 0.04f, nbr1, mask1);
  sa1_kernel<<<BATCH * N1, 32, 0, stream>>>(pos, c1, nbr1, mask1, w1a16, b1a, w1b16, b1b, x1);
  // SA2
  fps_kernel<<<BATCH, 256, 0, stream>>>(c1, N1, N2, idx2, c2);
  ballq_kernel<<<BATCH * N2 / 64, 64, 0, stream>>>(c2, c1, N2, N1, 0.16f, nbr2, mask2);
  sa2_kernel<<<BATCH * N2, 32, 0, stream>>>(c1, c2, nbr2, mask2, x1, w2a16, b2a, w2b16, b2b, x2);
  // Global SA + reduce
  sa3_kernel<<<BATCH * 8, 32, 0, stream>>>(c2, x2, w3a16, b3a, w3b16, b3b, partial);
  reduce_max_kernel<<<(BATCH * 1024 + 255) / 256, 256, 0, stream>>>(partial, out);
}
